// DisentangledGNN_26834955665983
// MI455X (gfx1250) — matrix operand order repaired and checked by reading, not compiled, not run on hardware
//
#include <hip/hip_runtime.h>
#include <math.h>

// DisentangledGNN forward for MI455X (gfx1250, wave32).
// Pipeline: WMMA GEMM1 (+bias+leaky) -> group L2 norm -> 3x (copy, edge-routing
// with 16 lanes/edge + shuffle reductions + global_atomic_add_f32, renorm)
// -> WMMA GEMM2 with leaky fused into the f16 A-operand conversion.
//
// Codegen notes: K-loops split into unguarded main body + single guarded tail;
// guards use clamped-address load + select (v_cndmask), never exec branches;
// A fragments load as float4 (global_load_b128).

typedef __attribute__((ext_vector_type(16))) _Float16 v16h;
typedef __attribute__((ext_vector_type(8)))  float    v8f;

#define NNODES  50000
#define NFEAT   500
#define KMAIN   480          // largest multiple of 32 <= NFEAT
#define DDIM    160
#define NCLASS  40
#define NEDGES  800000
#define KFAC    10
#define FDD     16
#define SLOPE   0.01f

__device__ __forceinline__ float leaky(float v) { return v > 0.f ? v : SLOPE * v; }

// Pack a float4 pair (8 contiguous K values) into a[base..base+7].
__device__ __forceinline__ void pack8(v16h& a, int base, float4 p0, float4 p1)
{
    a[base + 0] = (_Float16)p0.x; a[base + 1] = (_Float16)p0.y;
    a[base + 2] = (_Float16)p0.z; a[base + 3] = (_Float16)p0.w;
    a[base + 4] = (_Float16)p1.x; a[base + 5] = (_Float16)p1.y;
    a[base + 6] = (_Float16)p1.z; a[base + 7] = (_Float16)p1.w;
}

// ---------------------------------------------------------------------------
// GEMM1: out[50000,160] = leaky(x[50000,500] @ w[500,160] + bias), one wave
// per 16x16 tile, v_wmma_f32_16x16x32_f16, K chunks of 32.
// A layout (ISA 7.12.2, 16-bit A 16x32): lane M = lane&15, half h = lane>>4;
//   elems 0..7  -> K = kc + 8h + e, elems 8..15 -> K = kc + 16 + 8h + (e-8).
// B layout (32x16): lane N = lane&15; same K order as A (kc + 8h + e, kc + 16 + 8h + (e-8)).
// D layout: VGPR v, lane l -> M = v + 8*(l>>4), N = l&15.
// ---------------------------------------------------------------------------
__global__ __launch_bounds__(256)
void gemm1_wmma(const float* __restrict__ x, const float* __restrict__ w,
                const float* __restrict__ bias, float* __restrict__ out)
{
    const int NT = DDIM / 16;                               // 10 N-tiles
    const int wave = (blockIdx.x * blockDim.x + threadIdx.x) >> 5;
    if (wave >= (NNODES / 16) * NT) return;
    const int mt = wave / NT, nt = wave % NT;
    const int lane = threadIdx.x & 31;
    const int h = lane >> 4, r = lane & 15;

    const float* xrow = x + (size_t)(mt * 16 + r) * NFEAT;  // A row for this lane
    const int ncol = nt * 16 + r;
    const float* wcol = w + ncol;                           // B column for this lane

    v8f c = {};

    // ---- main body: kc in [0, 480), fully unguarded, b128 A loads ----------
    for (int kc = 0; kc < KMAIN; kc += 32) {
        const float* pa = xrow + kc + h * 8;
        float4 x0 = *(const float4*)(pa);
        float4 x1 = *(const float4*)(pa + 4);
        float4 x2 = *(const float4*)(pa + 16);
        float4 x3 = *(const float4*)(pa + 20);
        v16h a;
        pack8(a, 0, x0, x1);
        pack8(a, 8, x2, x3);

        v16h b;
        const float* pb = wcol + (size_t)(kc + h * 8) * DDIM;
#pragma unroll
        for (int e = 0; e < 16; ++e)
            b[e] = (_Float16)pb[(size_t)((e < 8) ? e : (e + 8)) * DDIM];

        c = __builtin_amdgcn_wmma_f32_16x16x32_f16(false, a, false, b,
                                                   (short)0, c, false, false);
    }

    // ---- tail chunk: kc = 480, 20 valid K; clamp+select (no branches) ------
    {
        v16h a;
#pragma unroll
        for (int e = 0; e < 16; ++e) {
            int k = KMAIN + h * 8 + ((e < 8) ? e : (8 + e));
            int kcl = (k < NFEAT) ? k : (NFEAT - 1);        // in-bounds address
            float v = xrow[kcl];
            a[e] = (_Float16)((k < NFEAT) ? v : 0.f);       // v_cndmask, no exec
        }
        v16h b;
#pragma unroll
        for (int e = 0; e < 16; ++e) {
            int k = KMAIN + h * 8 + ((e < 8) ? e : (8 + e));
            int kcl = (k < NFEAT) ? k : (NFEAT - 1);
            float v = wcol[(size_t)kcl * DDIM];
            b[e] = (_Float16)((k < NFEAT) ? v : 0.f);
        }
        c = __builtin_amdgcn_wmma_f32_16x16x32_f16(false, a, false, b,
                                                   (short)0, c, false, false);
    }

#pragma unroll
    for (int v = 0; v < 8; ++v) {
        int m = mt * 16 + v + 8 * h;
        out[(size_t)m * DDIM + ncol] = leaky(c[v] + bias[ncol]);
    }
}

// ---------------------------------------------------------------------------
// GEMM2: out[50000,40] = leaky(u) @ clf_w[160,40] + clf_b. K=160 = 5 clean
// chunks; leaky fused into A conversion; ncol>=40 lanes use clamp+select.
// ---------------------------------------------------------------------------
__global__ __launch_bounds__(256)
void gemm2_wmma(const float* __restrict__ u, const float* __restrict__ w,
                const float* __restrict__ bias, float* __restrict__ out)
{
    const int NT = 3;
    const int wave = (blockIdx.x * blockDim.x + threadIdx.x) >> 5;
    if (wave >= (NNODES / 16) * NT) return;
    const int mt = wave / NT, nt = wave % NT;
    const int lane = threadIdx.x & 31;
    const int h = lane >> 4, r = lane & 15;

    const float* urow = u + (size_t)(mt * 16 + r) * DDIM;
    const int ncol = nt * 16 + r;
    const bool nvalid = (ncol < NCLASS);
    const float* wcol = w + (nvalid ? ncol : 0);            // clamped column

    v8f c = {};
    for (int kc = 0; kc < DDIM; kc += 32) {
        const float* pa = urow + kc + h * 8;
        float4 x0 = *(const float4*)(pa);
        float4 x1 = *(const float4*)(pa + 4);
        float4 x2 = *(const float4*)(pa + 16);
        float4 x3 = *(const float4*)(pa + 20);
        x0.x = leaky(x0.x); x0.y = leaky(x0.y); x0.z = leaky(x0.z); x0.w = leaky(x0.w);
        x1.x = leaky(x1.x); x1.y = leaky(x1.y); x1.z = leaky(x1.z); x1.w = leaky(x1.w);
        x2.x = leaky(x2.x); x2.y = leaky(x2.y); x2.z = leaky(x2.z); x2.w = leaky(x2.w);
        x3.x = leaky(x3.x); x3.y = leaky(x3.y); x3.z = leaky(x3.z); x3.w = leaky(x3.w);
        v16h a;
        pack8(a, 0, x0, x1);
        pack8(a, 8, x2, x3);

        v16h b;
        const float* pb = wcol + (size_t)(kc + h * 8) * NCLASS;
#pragma unroll
        for (int e = 0; e < 16; ++e) {
            float v = pb[(size_t)((e < 8) ? e : (e + 8)) * NCLASS];               // always in-bounds
            b[e] = (_Float16)(nvalid ? v : 0.f);            // select, no branch
        }
        c = __builtin_amdgcn_wmma_f32_16x16x32_f16(false, a, false, b,
                                                   (short)0, c, false, false);
    }
    if (!nvalid) return;                                    // lanes with ncol >= 40 exit (divergent)
#pragma unroll
    for (int v = 0; v < 8; ++v) {
        int m = mt * 16 + v + 8 * h;
        out[(size_t)m * NCLASS + ncol] = c[v] + bias[ncol];
    }
}

// ---------------------------------------------------------------------------
// Per-node group L2 norm: 16 lanes per node (2 nodes/wave32); lane l owns
// element l of each of the 10 factor chunks; shfl_xor reduction stays inside
// each 16-lane half of the wave.
// ---------------------------------------------------------------------------
__global__ __launch_bounds__(256)
void group_l2norm(const float* __restrict__ in, float* __restrict__ out)
{
    const int gid = blockIdx.x * blockDim.x + threadIdx.x;
    const int node = gid >> 4;
    const int l = threadIdx.x & 15;
    if (node >= NNODES) return;
    const float* row = in + (size_t)node * DDIM;
    float* orow = out + (size_t)node * DDIM;
#pragma unroll
    for (int k = 0; k < KFAC; ++k) {
        float v = row[k * FDD + l];
        float s = v * v;
        s += __shfl_xor(s, 8);
        s += __shfl_xor(s, 4);
        s += __shfl_xor(s, 2);
        s += __shfl_xor(s, 1);
        orow[k * FDD + l] = v / fmaxf(sqrtf(s), 1e-12f);
    }
}

__global__ __launch_bounds__(256)
void copy_f4(const float4* __restrict__ s, float4* __restrict__ d, int n4)
{
    int i = blockIdx.x * blockDim.x + threadIdx.x;
    if (i < n4) d[i] = s[i];
}

// ---------------------------------------------------------------------------
// Edge routing: 16 lanes per edge. Per factor k: coalesced 64B gathers of
// xn[src] and u[trg], shuffle-reduced dot; softmax over the 10 factors
// (replicated per lane); coalesced global_atomic_add_f32 scatter into unext.
// ---------------------------------------------------------------------------
__global__ __launch_bounds__(256)
void edge_route(const float* __restrict__ xn, const float* __restrict__ u,
                float* __restrict__ unext,
                const int* __restrict__ srcIdx, const int* __restrict__ trgIdx)
{
    const int gid = blockIdx.x * blockDim.x + threadIdx.x;
    const int e = gid >> 4;
    const int l = threadIdx.x & 15;
    if (e >= NEDGES) return;
    const int s = srcIdx[e];
    const int t = trgIdx[e];
    const float* zrow = xn + (size_t)s * DDIM;
    const float* urow = u + (size_t)t * DDIM;

    float z[KFAC], dot[KFAC];
#pragma unroll
    for (int k = 0; k < KFAC; ++k) {
        float zv = zrow[k * FDD + l];
        float uv = urow[k * FDD + l];
        z[k] = zv;
        float d = zv * uv;
        d += __shfl_xor(d, 8);
        d += __shfl_xor(d, 4);
        d += __shfl_xor(d, 2);
        d += __shfl_xor(d, 1);
        dot[k] = d;                     // full factor dot on all 16 lanes
    }
    float mx = dot[0];
#pragma unroll
    for (int k = 1; k < KFAC; ++k) mx = fmaxf(mx, dot[k]);
    float p[KFAC], sum = 0.f;
#pragma unroll
    for (int k = 0; k < KFAC; ++k) { p[k] = __expf(dot[k] - mx); sum += p[k]; }
    const float inv = 1.f / sum;

    float* nrow = unext + (size_t)t * DDIM;
#pragma unroll
    for (int k = 0; k < KFAC; ++k)
        atomicAdd(&nrow[k * FDD + l], z[k] * p[k] * inv);
}

// ---------------------------------------------------------------------------
extern "C" void kernel_launch(void* const* d_in, const int* in_sizes, int n_in,
                              void* d_out, int out_size, void* d_ws, size_t ws_size,
                              hipStream_t stream)
{
    const float* x     = (const float*)d_in[0];
    const int*   edges = (const int*)d_in[1];
    const float* pca_w = (const float*)d_in[2];
    const float* pca_b = (const float*)d_in[3];
    const float* clf_w = (const float*)d_in[4];
    const float* clf_b = (const float*)d_in[5];
    float* out = (float*)d_out;

    const int* srcIdx = edges;              // edge_index[0]
    const int* trgIdx = edges + NEDGES;     // edge_index[1]

    // Workspace: 3 node buffers of 50000*160 f32 = 32 MB each (96 MB total).
    const size_t bufElems = (size_t)NNODES * DDIM;
    float* XN = (float*)d_ws;               // normalized input features (constant z source)
    float* U  = XN + bufElems;              // current u (also h0 temp)
    float* UN = U + bufElems;               // accumulator u_next

    // h0 = leaky(x @ pca_w + b) -> U
    {
        const int waves = (NNODES / 16) * (DDIM / 16);
        gemm1_wmma<<<(waves * 32 + 255) / 256, 256, 0, stream>>>(x, pca_w, pca_b, U);
    }
    // XN = group_l2norm(h0)
    group_l2norm<<<(NNODES * 16 + 255) / 256, 256, 0, stream>>>(U, XN);

    const int n4 = (int)(bufElems / 4);
    for (int it = 0; it < 3; ++it) {
        copy_f4<<<(n4 + 255) / 256, 256, 0, stream>>>((const float4*)XN, (float4*)UN, n4);
        const float* ucur = (it == 0) ? XN : U;     // u starts equal to x (= XN)
        edge_route<<<(NEDGES * 16 + 255) / 256, 256, 0, stream>>>(XN, ucur, UN, srcIdx, trgIdx);
        group_l2norm<<<(NNODES * 16 + 255) / 256, 256, 0, stream>>>(UN, U);
    }

    // out = leaky(U) @ clf_w + clf_b   (leaky fused into A conversion)
    {
        const int waves = (NNODES / 16) * 3;
        gemm2_wmma<<<(waves * 32 + 255) / 256, 256, 0, stream>>>(U, clf_w, clf_b, out);
    }
}
